// DeepTransitionRNN_31928786878509
// MI455X (gfx1250) — compile-verified, matrix operand order checked by reading, not run on hardware
//
#include <hip/hip_runtime.h>
#include <hip/hip_bf16.h>

// DeepTransitionRNN for MI455X (gfx1250).
// Persistent cooperative kernel: 8 WGs x 512 threads (16 wave32 waves each),
// 1 WMMA 16x16 output tile per wave. h exchanged via ping-pong bf16 buffer in
// L2 with a software device barrier. All GEMMs via v_wmma_f32_16x16x32_bf16.
// Round 5: laundered weight base pointers are typed as address_space(1) so the
// fragment loads lower to global_load_b128 (LOADcnt only) instead of
// flat_load_b128 (which also consumes DScnt and the LDS path).

#define T_STEPS 512
#define BATCH   128
#define DIM     256
#define HID     256
#define NLAYER  4
#define NWG     8
#define THREADS 512
#define CAT_PITCH 1040   // bytes per LDS row: 512 bf16 * 2 + 16 pad (bank-conflict-free b128)

typedef __attribute__((ext_vector_type(16))) __bf16       v16bf;
typedef __attribute__((ext_vector_type(8)))  float        v8f;
typedef __attribute__((ext_vector_type(4)))  unsigned int v4u;

// global-address-space pointer types (keep global_load after asm laundering)
typedef const char __attribute__((address_space(1)))* gcp;
typedef const v4u  __attribute__((address_space(1)))* gv4u;

union ABf { v16bf v; v4u q[2]; };

// packed weight offsets in bf16 ELEMENTS (x2 = bytes)
#define OFF_WR 0u
#define OFF_WZ 131072u
#define OFF_WL 262144u
#define OFF_WT 393216u
#define OFF_CX 458752u
#define OFF_CH 524288u
#define OFF_TR 589824u
#define OFF_TZ 851968u
#define OFF_TN 1114112u
#define W_ELEMS 1376256u

// byte offsets for the fragment loads (compile-time immediates)
#define BOFF_WR 0
#define BOFF_WZ 262144
#define BOFF_WL 524288
#define BOFF_WT 786432
#define BOFF_CX 917504
#define BOFF_CH 1048576
#define BOFF_TR 1179648
#define BOFF_TZ 1703936
#define BOFF_TN 2228224

// Launder a pointer through opaque asm: kills loop-invariance / CSE so the
// loads stay inside the loop body (zero instructions emitted).
#define OPAQUE(p) asm volatile("" : "+v"(p))

__device__ __forceinline__ unsigned short f2bf(float f) {
  unsigned u = __float_as_uint(f);
  u += 0x7FFFu + ((u >> 16) & 1u);          // round-to-nearest-even
  return (unsigned short)(u >> 16);
}
__device__ __forceinline__ float sigf(float x) {
  float e = __expf(-x);
  return __builtin_amdgcn_rcpf(1.0f + e);
}
__device__ __forceinline__ float tanhfast(float x) {
  float xc = fminf(15.0f, fmaxf(-15.0f, x));
  float e = __expf(2.0f * xc);
  return (e - 1.0f) * __builtin_amdgcn_rcpf(e + 1.0f);
}
__device__ __forceinline__ v8f wmma_bf16(const ABf& a, const ABf& b, v8f c) {
  return __builtin_amdgcn_wmma_f32_16x16x32_bf16(false, a.v, false, b.v,
                                                 (short)0, c, false, false);
}
// A fragment (16x32 bf16) from LDS: two ds_load_b128 at immediate offsets.
__device__ __forceinline__ void loadAo(ABf& a, const char* abase, int byteOff) {
  const v4u* p = (const v4u*)(abase + byteOff);
  a.q[0] = p[0];        // K = k0..k0+7
  a.q[1] = p[2];        // K = k0+16..k0+23  (+32 bytes)
}
// B fragment (32x16 bf16) from pre-packed weights: two global_load_b128,
// base pointer is global-AS + laundered, byteOff is a compile-time immediate.
__device__ __forceinline__ void loadBo(ABf& b, gcp base, int byteOff) {
  gv4u p = (gv4u)(base + byteOff);
  b.q[0] = p[0];
  b.q[1] = p[1];
}

// ---- pipeline stage macros (ping/pong fragment sets, offsets are immediates)
#define CELL_LOADS_H1(S, kc)                          \
  loadAo(S##A, abase, (kc) * 64);                     \
  loadBo(S##0, base16, BOFF_WR + (kc) * 1024);        \
  loadBo(S##1, base16, BOFF_WZ + (kc) * 1024);        \
  loadBo(S##2, base16, BOFF_WL + (kc) * 1024);        \
  loadBo(S##3, base8,  BOFF_CX + (kc) * 1024);        \
  loadBo(S##4, base8,  BOFF_WT + (kc) * 1024);

#define CELL_WMMA_H1(S)                               \
  aR  = wmma_bf16(S##A, S##0, aR);                    \
  aZ  = wmma_bf16(S##A, S##1, aZ);                    \
  aL  = wmma_bf16(S##A, S##2, aL);                    \
  aCx = wmma_bf16(S##A, S##3, aCx);                   \
  aWt = wmma_bf16(S##A, S##4, aWt);                   \
  __builtin_amdgcn_sched_barrier(0);

#define CELL_LOADS_H2(S, kh)                          \
  loadAo(S##A, abase, (8 + (kh)) * 64);               \
  loadBo(S##0, base16, BOFF_WR + (8 + (kh)) * 1024);  \
  loadBo(S##1, base16, BOFF_WZ + (8 + (kh)) * 1024);  \
  loadBo(S##2, base16, BOFF_WL + (8 + (kh)) * 1024);  \
  loadBo(S##3, base8,  BOFF_CH + (kh) * 1024);

#define CELL_WMMA_H2(S)                               \
  aR  = wmma_bf16(S##A, S##0, aR);                    \
  aZ  = wmma_bf16(S##A, S##1, aZ);                    \
  aL  = wmma_bf16(S##A, S##2, aL);                    \
  aCh = wmma_bf16(S##A, S##3, aCh);                   \
  __builtin_amdgcn_sched_barrier(0);

#define TR_LOADS(S, li, kc)                                      \
  loadAo(S##A, abase, (8 + (kc)) * 64);                          \
  loadBo(S##0, base8, BOFF_TR + (li) * 131072 + (kc) * 1024);    \
  loadBo(S##1, base8, BOFF_TZ + (li) * 131072 + (kc) * 1024);    \
  loadBo(S##2, base8, BOFF_TN + (li) * 131072 + (kc) * 1024);

#define TR_WMMA(S)                                    \
  aRR = wmma_bf16(S##A, S##0, aRR);                   \
  aZZ = wmma_bf16(S##A, S##1, aZZ);                   \
  aNN = wmma_bf16(S##A, S##2, aNN);                   \
  __builtin_amdgcn_sched_barrier(0);

__device__ __forceinline__ void grid_barrier(unsigned* cnt, unsigned* gen) {
  __threadfence();
  __syncthreads();
  if (threadIdx.x == 0) {
    unsigned g = __hip_atomic_load(gen, __ATOMIC_ACQUIRE, __HIP_MEMORY_SCOPE_AGENT);
    unsigned a = __hip_atomic_fetch_add(cnt, 1u, __ATOMIC_ACQ_REL, __HIP_MEMORY_SCOPE_AGENT);
    if (a == NWG - 1u) {
      __hip_atomic_store(cnt, 0u, __ATOMIC_RELAXED, __HIP_MEMORY_SCOPE_AGENT);
      __hip_atomic_fetch_add(gen, 1u, __ATOMIC_RELEASE, __HIP_MEMORY_SCOPE_AGENT);
    } else {
      while (__hip_atomic_load(gen, __ATOMIC_ACQUIRE, __HIP_MEMORY_SCOPE_AGENT) == g)
        __builtin_amdgcn_s_sleep(1);
    }
  }
  __syncthreads();
  __threadfence();
}

// cooperative: copy full h (128x256 bf16) from global ping-pong buffer to LDS cat[:,256:512]
__device__ __forceinline__ void loadH_to_lds(char* lds, const unsigned short* hb, int tid) {
  for (int e8 = tid; e8 < BATCH * 32; e8 += THREADS) {
    int r = e8 >> 5, c8 = e8 & 31;
    v4u val = ((const v4u*)hb)[e8];
    *(v4u*)(lds + r * CAT_PITCH + 512 + c8 * 16) = val;
  }
  __syncthreads();
}

__global__ __launch_bounds__(THREADS, 1)
void rnn_persistent_kernel(const unsigned short* __restrict__ xb,
                           const int* __restrict__ lengths,
                           const unsigned short* __restrict__ wts,
                           unsigned short* __restrict__ hbuf,   // 2 x 32768 bf16
                           float* __restrict__ out,
                           unsigned* __restrict__ bar) {
  extern __shared__ char lds[];
  const int tid   = threadIdx.x;
  const int lane  = tid & 31;
  const int wave  = tid >> 5;
  const int ln15  = lane & 15;
  const int khalf = lane >> 4;
  const int mTile = wave & 7;
  const int nTile = blockIdx.x * 2 + (wave >> 3);
  const int arow  = mTile * 16 + ln15;           // A-fragment row this lane feeds
  const int colG  = nTile * 16 + ln15;           // C-layout column this lane owns

  // per-wave fragment base pointers in the GLOBAL address space
  gcp gwts = (gcp)(unsigned long long)wts;
  gcp wbase16 = gwts + nTile * 16384 + lane * 32; // K=512 mats
  gcp wbase8  = gwts + nTile * 8192  + lane * 32; // K=256 mats
  const char* abase = lds + arow * CAT_PITCH + khalf * 16;  // A frags (LDS)

  // zero the h half of the cat buffer (h0 = 0)
  for (int e8 = tid; e8 < BATCH * 32; e8 += THREADS) {
    int r = e8 >> 5, c8 = e8 & 31;
    v4u z = {0u, 0u, 0u, 0u};
    *(v4u*)(lds + r * CAT_PITCH + 512 + c8 * 16) = z;
  }

  int rows[8], lenv[8];
#pragma unroll
  for (int v = 0; v < 8; ++v) {
    rows[v] = mTile * 16 + (khalf ? v + 8 : v);
    lenv[v] = lengths[rows[v]];
  }

  v8f h_reg = {0.f, 0.f, 0.f, 0.f, 0.f, 0.f, 0.f, 0.f};
  int pp = 0;
  __syncthreads();

  for (int t = 0; t < T_STEPS; ++t) {
    // stage xt (bf16, row-major) into cat[:,0:256]
    const unsigned short* xt = xb + (size_t)t * BATCH * DIM;
    for (int e8 = tid; e8 < BATCH * 32; e8 += THREADS) {
      int r = e8 >> 5, c8 = e8 & 31;
      v4u val = ((const v4u*)xt)[e8];
      *(v4u*)(lds + r * CAT_PITCH + c8 * 16) = val;
    }
    __syncthreads();

    v8f hold = h_reg;

    // ===== cell: r,z,l = sig(cat@W*), n = tanh(x@Cx + r*(h@Ch)) + l*(x@Wt) =====
    v8f aR  = {0.f,0.f,0.f,0.f,0.f,0.f,0.f,0.f};
    v8f aZ  = aR, aL = aR, aCx = aR, aCh = aR, aWt = aR;

    {   // K chunks 0..7 (xt half): Wr,Wz,Wl,Cx,Wt — ping-pong pipeline
      gcp base16 = wbase16;  OPAQUE(base16);
      gcp base8  = wbase8;   OPAQUE(base8);
      ABf PA, P0, P1, P2, P3, P4;
      ABf QA, Q0, Q1, Q2, Q3, Q4;
      CELL_LOADS_H1(P, 0);
      __builtin_amdgcn_sched_barrier(0);
      CELL_LOADS_H1(Q, 1);  CELL_WMMA_H1(P);
      CELL_LOADS_H1(P, 2);  CELL_WMMA_H1(Q);
      CELL_LOADS_H1(Q, 3);  CELL_WMMA_H1(P);
      CELL_LOADS_H1(P, 4);  CELL_WMMA_H1(Q);
      CELL_LOADS_H1(Q, 5);  CELL_WMMA_H1(P);
      CELL_LOADS_H1(P, 6);  CELL_WMMA_H1(Q);
      CELL_LOADS_H1(Q, 7);  CELL_WMMA_H1(P);
      CELL_WMMA_H1(Q);
    }
    {   // K chunks 8..15 (h half): Wr,Wz,Wl,Ch — ping-pong pipeline
      gcp base16 = wbase16;  OPAQUE(base16);
      gcp base8  = wbase8;   OPAQUE(base8);
      ABf PA, P0, P1, P2, P3;
      ABf QA, Q0, Q1, Q2, Q3;
      CELL_LOADS_H2(P, 0);
      __builtin_amdgcn_sched_barrier(0);
      CELL_LOADS_H2(Q, 1);  CELL_WMMA_H2(P);
      CELL_LOADS_H2(P, 2);  CELL_WMMA_H2(Q);
      CELL_LOADS_H2(Q, 3);  CELL_WMMA_H2(P);
      CELL_LOADS_H2(P, 4);  CELL_WMMA_H2(Q);
      CELL_LOADS_H2(Q, 5);  CELL_WMMA_H2(P);
      CELL_LOADS_H2(P, 6);  CELL_WMMA_H2(Q);
      CELL_LOADS_H2(Q, 7);  CELL_WMMA_H2(P);
      CELL_WMMA_H2(Q);
    }
#pragma unroll
    for (int v = 0; v < 8; ++v) {
      float r = sigf(aR[v]);
      float z = sigf(aZ[v]);
      float l = sigf(aL[v]);
      float n = tanhfast(aCx[v] + r * aCh[v]) + l * aWt[v];
      h_reg[v] = (1.0f - z) * hold[v] + z * n;
    }
    {
      unsigned short* hb = hbuf + pp * (BATCH * HID);
#pragma unroll
      for (int v = 0; v < 8; ++v) hb[rows[v] * HID + colG] = f2bf(h_reg[v]);
      grid_barrier(&bar[0], &bar[32]);
      loadH_to_lds(lds, hb, tid);
      pp ^= 1;
    }

    // ===== deep transition layers (li unrolled -> immediate offsets) =====
#pragma unroll
    for (int li = 0; li < NLAYER; ++li) {
      v8f aRR = {0.f,0.f,0.f,0.f,0.f,0.f,0.f,0.f};
      v8f aZZ = aRR, aNN = aRR;
      {
        gcp base8 = wbase8;  OPAQUE(base8);
        ABf PA, P0, P1, P2;
        ABf QA, Q0, Q1, Q2;
        TR_LOADS(P, li, 0);
        __builtin_amdgcn_sched_barrier(0);
        TR_LOADS(Q, li, 1);  TR_WMMA(P);
        TR_LOADS(P, li, 2);  TR_WMMA(Q);
        TR_LOADS(Q, li, 3);  TR_WMMA(P);
        TR_LOADS(P, li, 4);  TR_WMMA(Q);
        TR_LOADS(Q, li, 5);  TR_WMMA(P);
        TR_LOADS(P, li, 6);  TR_WMMA(Q);
        TR_LOADS(Q, li, 7);  TR_WMMA(P);
        TR_WMMA(Q);
      }
#pragma unroll
      for (int v = 0; v < 8; ++v) {
        float rr = sigf(aRR[v]);
        float zz = sigf(aZZ[v]);
        float nn = tanhfast(rr * aNN[v]);
        h_reg[v] = (1.0f - zz) * nn + zz * h_reg[v];
      }
      if (li < NLAYER - 1) {
        unsigned short* hb = hbuf + pp * (BATCH * HID);
#pragma unroll
        for (int v = 0; v < 8; ++v) hb[rows[v] * HID + colG] = f2bf(h_reg[v]);
        grid_barrier(&bar[0], &bar[32]);
        loadH_to_lds(lds, hb, tid);
        pp ^= 1;
      }
    }

    // ===== length mask + output, publish final h for next step =====
    float* ot = out + (size_t)t * (BATCH * HID);
#pragma unroll
    for (int v = 0; v < 8; ++v) {
      bool m = t < lenv[v];
      ot[rows[v] * HID + colG] = m ? h_reg[v] : 0.0f;
      h_reg[v] = m ? h_reg[v] : hold[v];
    }
    {
      unsigned short* hb = hbuf + pp * (BATCH * HID);
#pragma unroll
      for (int v = 0; v < 8; ++v) hb[rows[v] * HID + colG] = f2bf(h_reg[v]);
      grid_barrier(&bar[0], &bar[32]);
      loadH_to_lds(lds, hb, tid);
      pp ^= 1;
    }
  }
}

// ---- prep: pack fp32 weights (KxN row-major) into WMMA bf16 B-fragment layout ----
__global__ void pack_weight_kernel(const float* __restrict__ src,
                                   unsigned short* __restrict__ dst, int Ktot) {
  int total = Ktot * 256;
  int kcm = Ktot >> 5;
  for (int e = blockIdx.x * blockDim.x + threadIdx.x; e < total;
       e += gridDim.x * blockDim.x) {
    int j = e & 15;
    int lane = (e >> 4) & 31;
    int chunk = e >> 9;                 // nTile*kcm + kChunk
    int nTile = chunk / kcm;
    int kChunk = chunk - nTile * kcm;
    int n = nTile * 16 + (lane & 15);
    int khalf = lane >> 4;
    int k = kChunk * 32 + (j < 8 ? khalf * 8 + j : 16 + khalf * 8 + (j - 8));
    dst[e] = f2bf(src[k * 256 + n]);
  }
}

__global__ void cvt_x_kernel(const float4* __restrict__ src,
                             ushort4* __restrict__ dst, int n4) {
  int i = blockIdx.x * blockDim.x + threadIdx.x;
  if (i < n4) {
    float4 f = src[i];
    ushort4 o;
    o.x = f2bf(f.x); o.y = f2bf(f.y); o.z = f2bf(f.z); o.w = f2bf(f.w);
    dst[i] = o;
  }
}

extern "C" void kernel_launch(void* const* d_in, const int* in_sizes, int n_in,
                              void* d_out, int out_size, void* d_ws, size_t ws_size,
                              hipStream_t stream) {
  (void)in_sizes; (void)n_in; (void)out_size; (void)ws_size;
  const float* x       = (const float*)d_in[0];
  const int*   lengths = (const int*)d_in[1];
  const float* Wr = (const float*)d_in[2];
  const float* Wz = (const float*)d_in[3];
  const float* Wl = (const float*)d_in[4];
  const float* Wt = (const float*)d_in[5];
  const float* Cx = (const float*)d_in[6];
  const float* Ch = (const float*)d_in[7];
  const float* Tr = (const float*)d_in[8];
  const float* Tz = (const float*)d_in[9];
  const float* Tn = (const float*)d_in[10];

  char* ws = (char*)d_ws;
  unsigned*       bar  = (unsigned*)ws;                               // 256 B
  unsigned short* hbuf = (unsigned short*)(ws + 256);                 // 2*64 KB
  unsigned short* wts  = (unsigned short*)(ws + 256 + 131072);        // 2.75 MB
  unsigned short* xb   = (unsigned short*)(ws + 256 + 131072 + (size_t)W_ELEMS * 2);

  hipMemsetAsync(ws, 0, 256, stream);   // barrier state

  struct MatDesc { const float* src; int K; unsigned off; };
  MatDesc mats[18];
  mats[0] = {Wr, 512, OFF_WR};
  mats[1] = {Wz, 512, OFF_WZ};
  mats[2] = {Wl, 512, OFF_WL};
  mats[3] = {Wt, 256, OFF_WT};
  mats[4] = {Cx, 256, OFF_CX};
  mats[5] = {Ch, 256, OFF_CH};
  for (int i = 0; i < 4; ++i) {
    mats[6 + i]  = {Tr + (size_t)i * 65536, 256, OFF_TR + (unsigned)i * 65536u};
    mats[10 + i] = {Tz + (size_t)i * 65536, 256, OFF_TZ + (unsigned)i * 65536u};
    mats[14 + i] = {Tn + (size_t)i * 65536, 256, OFF_TN + (unsigned)i * 65536u};
  }
  for (int i = 0; i < 18; ++i) {
    int total = mats[i].K * 256;
    pack_weight_kernel<<<(total + 255) / 256, 256, 0, stream>>>(
        mats[i].src, wts + mats[i].off, mats[i].K);
  }
  int n4 = T_STEPS * BATCH * DIM / 4;
  cvt_x_kernel<<<(n4 + 255) / 256, 256, 0, stream>>>(
      (const float4*)x, (ushort4*)xb, n4);

  rnn_persistent_kernel<<<dim3(NWG), dim3(THREADS), BATCH * CAT_PITCH, stream>>>(
      xb, lengths, wts, hbuf, (float*)d_out, bar);
}
